// BahdanauDecoder_76982993814150
// MI455X (gfx1250) — matrix operand order repaired and checked by reading, not compiled, not run on hardware
//
#include <hip/hip_runtime.h>
#include <math.h>

// Problem dimensions (fixed by the reference)
constexpr int nB = 16, nT = 128, nS = 128, nH = 1024, nE = 512, nV = 32000;

typedef __attribute__((ext_vector_type(16))) __bf16 v16bf;
typedef __attribute__((ext_vector_type(8)))  float  v8f;

enum { ACT_NONE = 0, ACT_TANH = 1 };

// ---------------------------------------------------------------------------
// One-time weight pack: f32 weight -> bf16 in the exact WMMA B-fragment order.
// Fragment layout consumed by the GEMM below:
//   packed[(((nt*(K/32)) + kt)*32 + lane)*16 + j]
//     = W[k][n]   with k = kt*32 + (lane>>4)*16 + j,  n = nt*16 + (lane&15)
// transB=0: W stored K x N row-major (ld = N); transB=1: W stored N x K (ld = K).
// Each per-lane fragment is 32 contiguous bytes; a wave's fragment is one
// contiguous 1 KB block -> two global_load_b128 per fragment in the hot loop.
// ---------------------------------------------------------------------------
__global__ void k_pack_b(const float* __restrict__ src, __bf16* __restrict__ dst,
                         int K, int N, int ld, int transB)
{
    size_t i = (size_t)blockIdx.x * blockDim.x + threadIdx.x;
    if (i >= (size_t)K * N) return;
    int j    = (int)(i & 15);
    int lane = (int)((i >> 4) & 31);
    size_t f = i >> 9;                 // fragment index = nt*(K/32) + kt
    int kt   = (int)(f % (size_t)(K >> 5));
    int nt   = (int)(f / (size_t)(K >> 5));
    int k = kt * 32 + (lane >> 4) * 16 + j;
    int n = nt * 16 + (lane & 15);
    float v = transB ? src[(size_t)n * ld + k] : src[(size_t)k * ld + n];
    dst[i] = (__bf16)v;
}

// ---------------------------------------------------------------------------
// WMMA GEMM: C[M x N] = act(A[M x K] (f32) * Bpacked (bf16 fragments) + bias)
// One wave per 16x16 output tile; 4 waves per block along N.
// K multiple of 32, M and N multiples of 16 (true for every call here).
// ---------------------------------------------------------------------------
template <int ACT>
__global__ void k_gemm_bf16(const float* __restrict__ A, int lda,
                            const __bf16* __restrict__ Bp,
                            const float* __restrict__ bias,
                            float* __restrict__ C,
                            int M, int N, int K)
{
    const int lane  = threadIdx.x & 31;
    const int wave  = threadIdx.x >> 5;
    const int ntile = (blockIdx.x * 4 + wave) * 16;
    const int mtile = blockIdx.y * 16;
    if (ntile >= N || mtile >= M) return;   // wave-uniform: EXEC stays all-ones

    const int half = lane >> 4;   // 0: lanes 0-15, 1: lanes 16-31
    const int l15  = lane & 15;

    v8f acc = {};

    // A fragment base: row = mtile + l15; K sub-offset = half*16 (matches packed B)
    const float*  Arow  = A + (size_t)(mtile + l15) * lda + half * 16;
    // B fragments for this N-tile: contiguous 1 KB per K-step
    const __bf16* Bfrag = Bp + (size_t)(ntile >> 4) * (size_t)(K >> 5) * 512 + lane * 16;

    for (int k0 = 0; k0 < K; k0 += 32) {
        v16bf a, b;
        // A 16x32 bf16: elem j -> K = k0 + half*16 + j
        #pragma unroll
        for (int j = 0; j < 8; ++j) {
            a[j]     = (__bf16)Arow[k0 + j];
            a[j + 8] = (__bf16)Arow[k0 + 8 + j];
        }
        b = *(const v16bf*)Bfrag;          // 32B contiguous, 32B aligned
        Bfrag += 512;
        acc = __builtin_amdgcn_wmma_f32_16x16x32_bf16(
                  false, a, false, b, (short)0, acc, false, false);
    }

    // C/D layout: VGPR r -> M = mtile + r + half*8, N = ntile + l15
    const int n  = ntile + l15;
    const float bv = bias ? bias[n] : 0.0f;
    #pragma unroll
    for (int r = 0; r < 8; ++r) {
        const int m = mtile + r + half * 8;
        float v = acc[r] + bv;
        if (ACT == ACT_TANH) v = tanhf(v);
        C[(size_t)m * N + n] = v;
    }
}

// ---------------------------------------------------------------------------
// Elementwise / helper kernels
// ---------------------------------------------------------------------------

// ys[t][b][e] = emb[y_ids[b][t]][e]   (f32, laid out for per-step slicing)
__global__ void k_gather(const int* __restrict__ ids, const float* __restrict__ emb,
                         float* __restrict__ ys)
{
    int i = blockIdx.x * blockDim.x + threadIdx.x;
    if (i >= nT * nB * nE) return;
    int e  = i % nE;
    int bt = i / nE;
    int b  = bt % nB;
    int t  = bt / nB;
    int id = ids[b * nT + t];
    ys[i] = emb[(size_t)id * nE + e];
}

// energy[b][s] = mask ? dot(tanh(pq[b]+proj_key[b][s]), we) : -inf
// One wave per (b,s) pair; wave32 shuffle reduction.
__global__ void k_energy(const float* __restrict__ pq, const float* __restrict__ pk,
                         const float* __restrict__ we, const int* __restrict__ mask,
                         float* __restrict__ energy)
{
    int wid  = blockIdx.x * 4 + (threadIdx.x >> 5);
    int lane = threadIdx.x & 31;
    if (wid >= nB * nS) return;
    int b = wid / nS, s = wid % nS;
    const float* q = pq + (size_t)b * nH;
    const float* p = pk + ((size_t)b * nS + s) * nH;
    float sum = 0.0f;
    for (int i = lane; i < nH; i += 32)
        sum += tanhf(q[i] + p[i]) * we[i];
    #pragma unroll
    for (int off = 16; off; off >>= 1) sum += __shfl_xor(sum, off, 32);
    if (lane == 0)
        energy[wid] = (mask[b * nS + s] == 0) ? -__builtin_inff() : sum;
}

// context[b][d] = sum_s energy[b][s] * enc[b][s][d]   (reference uses raw energy!)
__global__ void k_context(const float* __restrict__ energy, const float* __restrict__ enc,
                          float* __restrict__ ctx)
{
    int i = blockIdx.x * blockDim.x + threadIdx.x;
    if (i >= nB * 2 * nH) return;
    int b = i / (2 * nH), d = i % (2 * nH);
    const float* e  = energy + b * nS;
    const float* eb = enc + ((size_t)b * nS) * (2 * nH) + d;
    float acc = 0.0f;
    for (int s = 0; s < nS; ++s) acc += e[s] * eb[(size_t)s * 2 * nH];
    ctx[i] = acc;
}

// x[b]   = [y_t[b] (E), ctx[b] (2H)]
// cat[b] = [h[b] (H), ctx[b] (2H), y_t[b] (E)]     (h = OLD hidden state)
__global__ void k_assemble(const float* __restrict__ yt, const float* __restrict__ ctx,
                           const float* __restrict__ h,
                           float* __restrict__ x, float* __restrict__ cat)
{
    const int XW = nE + 2 * nH;      // 2560
    const int CW = 3 * nH + nE;      // 3584
    int i = blockIdx.x * blockDim.x + threadIdx.x;
    if (i < nB * XW) {
        int b = i / XW, c = i % XW;
        x[i] = (c < nE) ? yt[b * nE + c] : ctx[b * 2 * nH + (c - nE)];
    } else {
        int j = i - nB * XW;
        if (j < nB * CW) {
            int b = j / CW, c = j % CW;
            float v;
            if (c < nH)          v = h[b * nH + c];
            else if (c < 3 * nH) v = ctx[b * 2 * nH + (c - nH)];
            else                 v = yt[b * nE + (c - 3 * nH)];
            cat[j] = v;
        }
    }
}

// GRU gate fusion; also writes dec_outputs[b][t][:] = h_new
__global__ void k_gru(const float* __restrict__ gi, const float* __restrict__ gh,
                      const float* __restrict__ h, float* __restrict__ hnew,
                      float* __restrict__ dec_out, int t)
{
    int i = blockIdx.x * blockDim.x + threadIdx.x;
    if (i >= nB * nH) return;
    int b = i / nH, c = i % nH;
    const float* gib = gi + (size_t)b * 3 * nH;
    const float* ghb = gh + (size_t)b * 3 * nH;
    float r = 1.0f / (1.0f + expf(-(gib[c] + ghb[c])));
    float z = 1.0f / (1.0f + expf(-(gib[nH + c] + ghb[nH + c])));
    float n = tanhf(gib[2 * nH + c] + r * ghb[2 * nH + c]);
    float hv = (1.0f - z) * n + z * h[i];
    hnew[i] = hv;
    dec_out[((size_t)b * nT + t) * nH + c] = hv;
}

// pre[b][j] = max(tt[b][2j], tt[b][2j+1])
__global__ void k_maxout(const float* __restrict__ tt, float* __restrict__ pre)
{
    int i = blockIdx.x * blockDim.x + threadIdx.x;
    if (i >= nB * nH) return;
    pre[i] = fmaxf(tt[2 * i], tt[2 * i + 1]);
}

// log_softmax over V per batch row + argmax -> predictions
__global__ __launch_bounds__(512)
void k_logsoftmax(const float* __restrict__ logits, float* __restrict__ out_logp,
                  int* __restrict__ out_pred, int t)
{
    __shared__ float smax[512];
    __shared__ int   sarg[512];
    __shared__ float ssum[512];
    int b = blockIdx.x, tid = threadIdx.x;
    const float* row = logits + (size_t)b * nV;

    float m = -__builtin_inff(); int arg = 0;
    for (int v = tid; v < nV; v += 512) {
        float x = row[v];
        if (x > m) { m = x; arg = v; }
    }
    smax[tid] = m; sarg[tid] = arg; __syncthreads();
    for (int off = 256; off; off >>= 1) {
        if (tid < off) {
            if (smax[tid + off] > smax[tid] ||
                (smax[tid + off] == smax[tid] && sarg[tid + off] < sarg[tid])) {
                smax[tid] = smax[tid + off]; sarg[tid] = sarg[tid + off];
            }
        }
        __syncthreads();
    }
    float rowmax = smax[0];

    float sum = 0.0f;
    for (int v = tid; v < nV; v += 512) sum += expf(row[v] - rowmax);
    ssum[tid] = sum; __syncthreads();
    for (int off = 256; off; off >>= 1) {
        if (tid < off) ssum[tid] += ssum[tid + off];
        __syncthreads();
    }
    float lse = rowmax + logf(ssum[0]);

    float* op = out_logp + ((size_t)b * nT + t) * nV;
    for (int v = tid; v < nV; v += 512) op[v] = row[v] - lse;
    if (tid == 0) out_pred[b * nT + t] = sarg[0];
}

// ---------------------------------------------------------------------------
// Host side
// ---------------------------------------------------------------------------
static void launch_gemm(hipStream_t s, const float* A, int lda, const __bf16* Bp,
                        const float* bias, float* C, int M, int N, int K, bool act_tanh)
{
    dim3 grid((N / 16 + 3) / 4, M / 16), blk(128);
    if (act_tanh) k_gemm_bf16<ACT_TANH><<<grid, blk, 0, s>>>(A, lda, Bp, bias, C, M, N, K);
    else          k_gemm_bf16<ACT_NONE><<<grid, blk, 0, s>>>(A, lda, Bp, bias, C, M, N, K);
}

extern "C" void kernel_launch(void* const* d_in, const int* in_sizes, int n_in,
                              void* d_out, int out_size, void* d_ws, size_t ws_size,
                              hipStream_t stream)
{
    (void)in_sizes; (void)n_in; (void)out_size; (void)ws_size;

    const int*   y_ids = (const int*)  d_in[0];
    const int*   mask  = (const int*)  d_in[1];
    const float* enc   = (const float*)d_in[2];   // (B,S,2H)
    const float* enc_h = (const float*)d_in[3];   // (1,B,H)
    const float* Wq    = (const float*)d_in[4];
    const float* bq    = (const float*)d_in[5];
    const float* Wk    = (const float*)d_in[6];
    const float* bk    = (const float*)d_in[7];
    const float* we    = (const float*)d_in[8];
    const float* Wi    = (const float*)d_in[9];
    const float* bi    = (const float*)d_in[10];
    const float* emb   = (const float*)d_in[11];
    const float* W_ih  = (const float*)d_in[12];  // (3H, E+2H)
    const float* W_hh  = (const float*)d_in[13];  // (3H, H)
    const float* b_ih  = (const float*)d_in[14];
    const float* b_hh  = (const float*)d_in[15];
    const float* Wm    = (const float*)d_in[16];  // (3H+E, 2H)
    const float* bm    = (const float*)d_in[17];
    const float* Wp    = (const float*)d_in[18];  // (H, V)
    const float* bp    = (const float*)d_in[19];

    // Output slices (concatenated flat, return order)
    float* dec_out  = (float*)d_out;                          // B*T*H
    float* logp_out = dec_out + (size_t)nB * nT * nH;         // B*T*V
    int*   pred_out = (int*)(logp_out + (size_t)nB * nT * nV);// B*T

    // Workspace carve-up (bf16 fragment-packed weights, ~108 MB -> L2-resident)
    char* w = (char*)d_ws;
    auto alloc = [&](size_t bytes) -> char* {
        char* p = w; w += (bytes + 255) & ~(size_t)255; return p;
    };
    __bf16* Wq_b  = (__bf16*)alloc((size_t)nH * nH * 2);
    __bf16* Wk_b  = (__bf16*)alloc((size_t)2 * nH * nH * 2);
    __bf16* Wi_b  = (__bf16*)alloc((size_t)nH * nH * 2);
    __bf16* Wih_b = (__bf16*)alloc((size_t)3 * nH * (nE + 2 * nH) * 2);
    __bf16* Whh_b = (__bf16*)alloc((size_t)3 * nH * nH * 2);
    __bf16* Wm_b  = (__bf16*)alloc((size_t)(3 * nH + nE) * 2 * nH * 2);
    __bf16* Wp_b  = (__bf16*)alloc((size_t)nH * nV * 2);
    float* ys     = (float*)alloc((size_t)nT * nB * nE * 4);   // [T][B][E]
    float* pkey   = (float*)alloc((size_t)nB * nS * nH * 4);
    float* hA     = (float*)alloc((size_t)nB * nH * 4);
    float* hBuf   = (float*)alloc((size_t)nB * nH * 4);
    float* pq     = (float*)alloc((size_t)nB * nH * 4);
    float* energy = (float*)alloc((size_t)nB * nS * 4);
    float* ctx    = (float*)alloc((size_t)nB * 2 * nH * 4);
    float* xbuf   = (float*)alloc((size_t)nB * (nE + 2 * nH) * 4);
    float* catbuf = (float*)alloc((size_t)nB * (3 * nH + nE) * 4);
    float* gi     = (float*)alloc((size_t)nB * 3 * nH * 4);
    float* gh     = (float*)alloc((size_t)nB * 3 * nH * 4);
    float* tt     = (float*)alloc((size_t)nB * 2 * nH * 4);
    float* pre    = (float*)alloc((size_t)nB * nH * 4);
    float* logits = (float*)alloc((size_t)nB * nV * 4);

    // One-time: pack weights to bf16 WMMA B-fragment layout
    auto pack = [&](const float* src, __bf16* dst, int K, int N, int ld, int transB) {
        size_t tot = (size_t)K * N;
        k_pack_b<<<dim3((unsigned)((tot + 255) / 256)), 256, 0, stream>>>(src, dst, K, N, ld, transB);
    };
    pack(Wq,   Wq_b,  nH,          nH,     nH,          0);
    pack(Wk,   Wk_b,  2 * nH,      nH,     nH,          0);
    pack(Wi,   Wi_b,  nH,          nH,     nH,          0);
    pack(W_ih, Wih_b, nE + 2 * nH, 3 * nH, nE + 2 * nH, 1);   // used as x @ W_ih^T
    pack(W_hh, Whh_b, nH,          3 * nH, nH,          1);   // used as h @ W_hh^T
    pack(Wm,   Wm_b,  3 * nH + nE, 2 * nH, 2 * nH,      0);
    pack(Wp,   Wp_b,  nH,          nV,     nV,          0);

    // One-time: embedding gather, h0, proj_key
    k_gather<<<dim3((nT * nB * nE + 255) / 256), 256, 0, stream>>>(y_ids, emb, ys);
    launch_gemm(stream, enc_h, nH, Wi_b, bi, hA, nB, nH, nH, true);                 // h0=tanh(.)
    launch_gemm(stream, enc, 2 * nH, Wk_b, bk, pkey, nB * nS, nH, 2 * nH, false);

    float* hcur = hA;
    float* hnxt = hBuf;

    for (int t = 0; t < nT; ++t) {
        const float* y_t = ys + (size_t)t * nB * nE;

        // Attention (softmax 'alphas' is dead code in the reference: skip it)
        launch_gemm(stream, hcur, nH, Wq_b, bq, pq, nB, nH, nH, false);
        k_energy<<<dim3((nB * nS + 3) / 4), 128, 0, stream>>>(pq, pkey, we, mask, energy);
        k_context<<<dim3((nB * 2 * nH + 255) / 256), 256, 0, stream>>>(energy, enc, ctx);

        // Concats (x for GRU input, cat uses OLD h)
        {
            int tot = nB * ((nE + 2 * nH) + (3 * nH + nE));
            k_assemble<<<dim3((tot + 255) / 256), 256, 0, stream>>>(y_t, ctx, hcur, xbuf, catbuf);
        }

        // GRU
        launch_gemm(stream, xbuf, nE + 2 * nH, Wih_b, b_ih, gi, nB, 3 * nH, nE + 2 * nH, false);
        launch_gemm(stream, hcur, nH, Whh_b, b_hh, gh, nB, 3 * nH, nH, false);
        k_gru<<<dim3((nB * nH + 255) / 256), 256, 0, stream>>>(gi, gh, hcur, hnxt, dec_out, t);

        // Maxout head + vocab projection + log-softmax/argmax
        launch_gemm(stream, catbuf, 3 * nH + nE, Wm_b, bm, tt, nB, 2 * nH, 3 * nH + nE, false);
        k_maxout<<<dim3((nB * nH + 255) / 256), 256, 0, stream>>>(tt, pre);
        launch_gemm(stream, pre, nH, Wp_b, bp, logits, nB, nV, nH, false);
        k_logsoftmax<<<dim3(nB), 512, 0, stream>>>(logits, logp_out, pred_out, t);

        // ping-pong hidden state
        float* tmp = hcur; hcur = hnxt; hnxt = tmp;
    }
}
